// Head_72834055405795
// MI455X (gfx1250) — matrix operand order repaired and checked by reading, not compiled, not run on hardware
//
#include <hip/hip_runtime.h>
#include <hip/hip_bf16.h>

// ---- problem constants (from reference) ----
#define BB 8
#define TT 2048
#define DD 384
#define KK 64

typedef __attribute__((ext_vector_type(16))) _Float16 half16;
typedef __attribute__((ext_vector_type(8)))  _Float16 half8;
typedef __attribute__((ext_vector_type(8)))  float    float8;

__device__ __forceinline__ float8 wmma_f16(half16 a, half16 b, float8 c) {
    return __builtin_amdgcn_wmma_f32_16x16x32_f16(
        /*neg_a=*/false, a, /*neg_b=*/false, b,
        /*c_mod=*/(short)0, c, /*reuse_a=*/false, /*reuse_b=*/false);
}

// A-matrix fragment, 16x32 f16, wave32 layout:
//   lane m = lane&15, hi = lane>>4
//   halves 0..7  -> K = kbase + hi*8 + j
//   halves 8..15 -> K = kbase + 16 + hi*8 + j
__device__ __forceinline__ half16 load_a_frag(const _Float16* base, int stride, int kbase) {
    const int lane = threadIdx.x & 31;
    const int m = lane & 15, hi = lane >> 4;
    const _Float16* row = base + (size_t)m * stride;
    half8 lo = *(const half8*)(row + kbase + hi * 8);
    half8 hh = *(const half8*)(row + kbase + 16 + hi * 8);
    half16 r;
#pragma unroll
    for (int i = 0; i < 8; ++i) { r[i] = lo[i]; r[i + 8] = hh[i]; }
    return r;
}

// B-matrix fragment, 32x16 f16 (K x N), from storage where element [K][n]
// lives at base[n*stride + K] (K contiguous).
//   lane n = lane&15, hi = lane>>4; halves 0..7 -> K = kbase + hi*8 + j,
//   halves 8..15 -> K = kbase + 16 + hi*8 + j
__device__ __forceinline__ half16 load_b_frag(const _Float16* base, int stride, int kbase) {
    const int lane = threadIdx.x & 31;
    const int n = lane & 15, hi = lane >> 4;
    const _Float16* p = base + (size_t)n * stride + kbase + hi * 8;
    half8 lo = *(const half8*)(p);
    half8 hh = *(const half8*)(p + 16);
    half16 r;
#pragma unroll
    for (int i = 0; i < 8; ++i) { r[i] = lo[i]; r[i + 8] = hh[i]; }
    return r;
}

// ---------------------------------------------------------------------------
// Kernel 0: one-time weight transpose + f16 convert.
//   wT[n*DD + k] = W[k*ncols + n]   (K contiguous per output column)
// ---------------------------------------------------------------------------
__global__ __launch_bounds__(256) void transpose_w_kernel(
    const float* __restrict__ W, _Float16* __restrict__ wT, int ncols)
{
    int idx = blockIdx.x * 256 + threadIdx.x;
    int total = ncols * DD;
    if (idx >= total) return;
    int n = idx / DD, k = idx - n * DD;
    wT[idx] = (_Float16)W[(size_t)k * ncols + n];
}

// ---------------------------------------------------------------------------
// Kernel 1: q = rope(x@Wq), k = rope(x@Wk), v = x@Wv   (f16 outputs to ws)
//   q,k stored row-major (B*T, 64); v stored transposed per batch (384, T).
// grid = B*T/16 blocks of 256 threads (8 waves). Each block owns 16 rows.
// ---------------------------------------------------------------------------
__global__ __launch_bounds__(256) void qkv_rope_kernel(
    const float* __restrict__ x, const _Float16* __restrict__ wTq,
    const _Float16* __restrict__ wTk, const _Float16* __restrict__ wTv,
    _Float16* __restrict__ qh, _Float16* __restrict__ kh,
    _Float16* __restrict__ vT)
{
    __shared__ _Float16 xs[16][DD + 8];   // row stride 392 halves = 784B (16B-aligned)

    const int tid = threadIdx.x;
    const int g0  = blockIdx.x * 16;      // global row base (tiles never straddle batches)
    const int b   = g0 / TT;
    const int t0  = g0 - b * TT;          // sequence-local row base

    // stage x tile as f16
    for (int i = tid; i < 16 * DD; i += 256) {
        int r = i / DD, c = i - r * DD;
        xs[r][c] = (_Float16)x[(size_t)(g0 + r) * DD + c];
    }
    __syncthreads();

    const int lane = tid & 31;
    const int wv   = tid >> 5;
    const int nl   = lane & 15, hi = lane >> 4;

    // A fragments are invariant across all n-tiles: keep all 12 in registers.
    half16 afr[DD / 32];
#pragma unroll
    for (int kc = 0; kc < DD / 32; ++kc)
        afr[kc] = load_a_frag(&xs[0][0], DD + 8, kc * 32);

    // 32 n-tiles total: 0-3 -> q, 4-7 -> k, 8-31 -> v
    for (int nt = wv; nt < 32; nt += 8) {
        const _Float16* Wt; int col0, mode;
        if (nt < 4)       { Wt = wTq; col0 = nt * 16;       mode = 0; }
        else if (nt < 8)  { Wt = wTk; col0 = (nt - 4) * 16; mode = 1; }
        else              { Wt = wTv; col0 = (nt - 8) * 16; mode = 2; }
        const _Float16* wbase = Wt + (size_t)col0 * DD;

        // 4-deep software pipeline: each WMMA consumes a B-fragment whose
        // load was issued 4 k-chunks earlier; replacement load issued before
        // the consuming WMMA so >=4 fragments stay in flight.
        half16 bfr[4];
#pragma unroll
        for (int i = 0; i < 4; ++i)
            bfr[i] = load_b_frag(wbase, DD, i * 32);

        float8 acc = {};
#pragma unroll
        for (int kc = 0; kc < DD / 32; ++kc) {
            half16 bcur = bfr[kc & 3];
            if (kc + 4 < DD / 32)
                bfr[kc & 3] = load_b_frag(wbase, DD, (kc + 4) * 32);
            acc = wmma_f16(afr[kc], bcur, acc);
        }

        if (mode < 2) {
            // RoPE on f32 accumulators. C/D layout: lane holds N = nl for rows
            // M = r + 8*hi. Pair partner (N^1) sits in lane^1.
            const int ncol = col0 + nl;                 // 0..63
            const float inv = __powf(10000.0f, -(float)(ncol & ~1) / (float)KK);
            _Float16* dst = (mode == 0) ? qh : kh;
#pragma unroll
            for (int r = 0; r < 8; ++r) {
                int m = r + hi * 8;
                float ang = (float)(t0 + m) * inv;
                float cs = __cosf(ang), sn = __sinf(ang);
                float v0 = acc[r];
                float v1 = __shfl_xor(v0, 1, 32);
                float out = (ncol & 1) ? (v0 * cs + v1 * sn)
                                       : (v0 * cs - v1 * sn);
                dst[(size_t)(g0 + m) * KK + ncol] = (_Float16)out;
            }
        } else {
            const int d = col0 + nl;
#pragma unroll
            for (int r = 0; r < 8; ++r) {
                int m = r + hi * 8;
                vT[((size_t)b * DD + d) * TT + t0 + m] = (_Float16)acc[r];
            }
        }
    }
}

// ---------------------------------------------------------------------------
// Kernel 2: causal flash attention. One wave per 16-query tile; online
// softmax over 32-key blocks; O accumulated in 24 f32 C-fragments.
// grid = B*T/(16*4) blocks of 128 threads (4 waves).
// ---------------------------------------------------------------------------
__global__ __launch_bounds__(128) void attn_kernel(
    const _Float16* __restrict__ qh, const _Float16* __restrict__ kh,
    const _Float16* __restrict__ vT, float* __restrict__ out)
{
    __shared__ _Float16 sP[4][16][48];    // per-wave P bounce; row stride 96B

    const int tid = threadIdx.x;
    const int wv  = tid >> 5;
    const int lane = tid & 31;
    const int nl = lane & 15, hi = lane >> 4;

    const int qt = blockIdx.x * 4 + wv;   // 0 .. B*T/16-1
    const int b  = qt >> 7;               // / (T/16)
    const int t0 = (qt & 127) << 4;       // sequence-local query base

    const _Float16* qb = qh + ((size_t)b * TT + t0) * KK;
    const _Float16* kb = kh + (size_t)b * TT * KK;
    const _Float16* vb = vT + (size_t)b * DD * TT;

    // Q fragments (16x64) cached in registers
    half16 qa0 = load_a_frag(qb, KK, 0);
    half16 qa1 = load_a_frag(qb, KK, 32);

    float8 O[DD / 16];
#pragma unroll
    for (int i = 0; i < DD / 16; ++i) O[i] = (float8){};
    float rm[8], rl[8];
#pragma unroll
    for (int i = 0; i < 8; ++i) { rm[i] = -1e30f; rl[i] = 0.0f; }

    const int nblocks = (t0 + 47) >> 5;   // keys 0 .. t0+15 in blocks of 32

    for (int sb = 0; sb < nblocks; ++sb) {
        const int s0 = sb * 32;

        // prefetch next key block (one 128B key row per lane)
        if (sb + 1 < nblocks)
            __builtin_prefetch((const void*)(kb + (size_t)(s0 + 32 + lane) * KK), 0, 1);

        // ---- scores: S = Q @ K^T; load all 4 K-frags, then 4 WMMAs ----
        const _Float16* kp0 = kb + (size_t)s0 * KK;
        const _Float16* kp1 = kp0 + 16 * KK;
        half16 kb00 = load_b_frag(kp0, KK, 0);
        half16 kb01 = load_b_frag(kp0, KK, 32);
        half16 kb10 = load_b_frag(kp1, KK, 0);
        half16 kb11 = load_b_frag(kp1, KK, 32);
        float8 S0 = {}, S1 = {};
        S0 = wmma_f16(qa0, kb00, S0);
        S0 = wmma_f16(qa1, kb01, S0);
        S1 = wmma_f16(qa0, kb10, S1);
        S1 = wmma_f16(qa1, kb11, S1);

        // ---- causal mask + online softmax (16-lane row reductions) ----
        float alpha[8];
#pragma unroll
        for (int r = 0; r < 8; ++r) {
            const int M = r + hi * 8;
            const int trow = t0 + M;
            float s0v = S0[r] * 0.125f;
            float s1v = S1[r] * 0.125f;
            if (s0 + nl      > trow) s0v = -1e30f;
            if (s0 + nl + 16 > trow) s1v = -1e30f;
            float mx = fmaxf(s0v, s1v);
#pragma unroll
            for (int d = 1; d < 16; d <<= 1) mx = fmaxf(mx, __shfl_xor(mx, d, 32));
            float mnew = fmaxf(rm[r], mx);
            float a_ = __expf(rm[r] - mnew);
            float p0 = __expf(s0v - mnew);
            float p1 = __expf(s1v - mnew);
            float sum = p0 + p1;
#pragma unroll
            for (int d = 1; d < 16; d <<= 1) sum += __shfl_xor(sum, d, 32);
            rl[r] = rl[r] * a_ + sum;
            rm[r] = mnew;
            alpha[r] = a_;
            sP[wv][M][nl]      = (_Float16)p0;
            sP[wv][M][nl + 16] = (_Float16)p1;
        }

        // rescale accumulators
#pragma unroll
        for (int nt = 0; nt < DD / 16; ++nt)
#pragma unroll
            for (int r = 0; r < 8; ++r) O[nt][r] *= alpha[r];

        // ---- O += P @ V, 4-deep pipelined V fragments ----
        half16 pa = load_a_frag(&sP[wv][0][0], 48, 0);
        half16 vfr[4];
#pragma unroll
        for (int i = 0; i < 4; ++i)
            vfr[i] = load_b_frag(vb + (size_t)(i * 16) * TT + s0, TT, 0);
#pragma unroll
        for (int nt = 0; nt < DD / 16; ++nt) {
            half16 vcur = vfr[nt & 3];
            if (nt + 4 < DD / 16)
                vfr[nt & 3] = load_b_frag(vb + (size_t)((nt + 4) * 16) * TT + s0, TT, 0);
            O[nt] = wmma_f16(pa, vcur, O[nt]);
        }
    }

    // ---- epilogue: divide by row sums, store f32 ----
    float* ob = out + ((size_t)b * TT + t0) * DD;
#pragma unroll
    for (int nt = 0; nt < DD / 16; ++nt) {
#pragma unroll
        for (int r = 0; r < 8; ++r) {
            int M = r + hi * 8;
            ob[(size_t)M * DD + nt * 16 + nl] = O[nt][r] / rl[r];
        }
    }
}

extern "C" void kernel_launch(void* const* d_in, const int* in_sizes, int n_in,
                              void* d_out, int out_size, void* d_ws, size_t ws_size,
                              hipStream_t stream) {
    (void)in_sizes; (void)n_in; (void)out_size; (void)ws_size;
    const float* x  = (const float*)d_in[0];
    const float* Wq = (const float*)d_in[1];
    const float* Wk = (const float*)d_in[2];
    const float* Wv = (const float*)d_in[3];

    const size_t qElems  = (size_t)BB * TT * KK;   // 1,048,576 halves
    const size_t vElems  = (size_t)BB * DD * TT;   // 6,291,456 halves
    const size_t wqkE    = (size_t)DD * KK;        // 24,576 halves
    const size_t wvE     = (size_t)DD * DD;        // 147,456 halves

    _Float16* qh  = (_Float16*)d_ws;
    _Float16* kh  = qh  + qElems;
    _Float16* vT  = kh  + qElems;
    _Float16* wTq = vT  + vElems;
    _Float16* wTk = wTq + wqkE;
    _Float16* wTv = wTk + wqkE;

    transpose_w_kernel<<<(int)((wqkE + 255) / 256), 256, 0, stream>>>(Wq, wTq, KK);
    transpose_w_kernel<<<(int)((wqkE + 255) / 256), 256, 0, stream>>>(Wk, wTk, KK);
    transpose_w_kernel<<<(int)((wvE  + 255) / 256), 256, 0, stream>>>(Wv, wTv, DD);

    qkv_rope_kernel<<<BB * TT / 16, 256, 0, stream>>>(x, wTq, wTk, wTv, qh, kh, vT);
    attn_kernel<<<BB * TT / (16 * 4), 128, 0, stream>>>(qh, kh, vT, (float*)d_out);
}
